// FullGraphBaseline_60189671686512
// MI455X (gfx1250) — matrix operand order repaired and checked by reading, not compiled, not run on hardware
//
#include <hip/hip_runtime.h>
#include <hip/hip_bf16.h>

typedef __attribute__((ext_vector_type(16))) __bf16 v16bf;
typedef __attribute__((ext_vector_type(8)))  __bf16 v8bf;
typedef __attribute__((ext_vector_type(8)))  float  v8f;

#define COS_EPS 1e-8f
#define SIM_THRESH 0.2f

constexpr int EPI_PLAIN = 0, EPI_BIAS_RELU = 1, EPI_SIM = 2;

// --------------------------------------------------------------------------
// CDNA5 async global->LDS copy path (ASYNCcnt), guarded so either toolchain
// compiles. Fallback: synchronous global->VGPR->ds_store staging.
// --------------------------------------------------------------------------
#ifndef __has_builtin
#define __has_builtin(x) 0
#endif

#if __has_builtin(__builtin_amdgcn_global_load_async_to_lds_b128)
#define ASYNC_LDS 1
#else
#define ASYNC_LDS 0
#endif

#if __has_builtin(__builtin_amdgcn_s_wait_asynccnt)
#define WAIT_ASYNC() __builtin_amdgcn_s_wait_asynccnt(0)
#elif ASYNC_LDS
#define WAIT_ASYNC() asm volatile("s_wait_asynccnt 0x0" ::: "memory")
#else
#define WAIT_ASYNC() ((void)0)
#endif

#if ASYNC_LDS
// builtin wants: (b128 vector ptr in AS(1), b128 vector ptr in AS(3), imm offset, imm cpol)
typedef int b128i_t __attribute__((vector_size(4 * sizeof(int))));
typedef __attribute__((address_space(1))) b128i_t* gbl_b128_p;
typedef __attribute__((address_space(3))) b128i_t* lds_b128_p;
__device__ __forceinline__ void async_cp16(__bf16* lds_dst, const __bf16* g_src) {
  gbl_b128_p gp = (gbl_b128_p)(uintptr_t)g_src;                 // global: same 64-bit value
  lds_b128_p lp = (lds_b128_p)(unsigned int)(uintptr_t)lds_dst; // LDS addr = low 32 bits
  __builtin_amdgcn_global_load_async_to_lds_b128(gp, lp, 0, 0);
}
#endif

// ---------------------------------------------------------------------------
// prep: per-row L2 norm -> nrm (bf16, row-major [N,D]) and hT0 = emb^T (bf16)
// ---------------------------------------------------------------------------
__global__ __launch_bounds__(256) void prep_kernel(const float* __restrict__ emb,
                                                   __bf16* __restrict__ nrm,
                                                   __bf16* __restrict__ hT) {
  __shared__ float red[256];
  const int row = blockIdx.x;  // 0..4095
  const int tid = threadIdx.x;
  const float* e = emb + (size_t)row * 768;
  float s = 0.f;
  for (int j = tid; j < 768; j += 256) { float v = e[j]; s += v * v; }
  red[tid] = s;
  __syncthreads();
  for (int st = 128; st > 0; st >>= 1) {
    if (tid < st) red[tid] += red[tid + st];
    __syncthreads();
  }
  const float inv = 1.f / fmaxf(sqrtf(red[0]), COS_EPS);
  for (int j = tid; j < 768; j += 256) {
    float v = e[j];
    nrm[(size_t)row * 768 + j] = (__bf16)(v * inv);
    hT[(size_t)j * 4096 + row] = (__bf16)v;
  }
}

// ---------------------------------------------------------------------------
// convert W1,W2,W3 (768x768 f32) to bf16, packed consecutively
// ---------------------------------------------------------------------------
__global__ __launch_bounds__(256) void convw_kernel(const float* __restrict__ W1,
                                                    const float* __restrict__ W2,
                                                    const float* __restrict__ W3,
                                                    __bf16* __restrict__ Wb) {
  const int S = 768 * 768;
  int i = blockIdx.x * 256 + threadIdx.x;
  if (i < S) {
    Wb[i]         = (__bf16)W1[i];
    Wb[S + i]     = (__bf16)W2[i];
    Wb[2 * S + i] = (__bf16)W3[i];
  }
}

// ---------------------------------------------------------------------------
// NT GEMM: C[M,N] = A[M,K] (row-major) x B[N,K]^T (row-major rows = columns)
// bf16 in, f32 WMMA accumulate, bf16 out. 128x128x32 tiles, 8 waves,
// double-buffered LDS with async global->LDS staging (ASYNCcnt pipeline).
// ---------------------------------------------------------------------------
template <int EPI>
__global__ __launch_bounds__(256) void gemm_nt_bf16(const __bf16* __restrict__ A,
                                                    const __bf16* __restrict__ B,
                                                    __bf16* __restrict__ C,
                                                    const float* __restrict__ bias,
                                                    int M, int N, int K) {
  constexpr int BM = 128, BN = 128, BK = 32;
  constexpr int LDT = BK + 8;  // padded LDS row: 80B = 5x16B, keeps v8bf aligned
  __shared__ __align__(16) __bf16 sA[2][BM * LDT];
  __shared__ __align__(16) __bf16 sB[2][BN * LDT];

  const int tid  = threadIdx.x;
  const int lane = tid & 31;
  const int wave = tid >> 5;
  const int wm   = wave >> 2;  // 0..1 : 64-row slab
  const int wn   = wave & 3;   // 0..3 : 32-col slab
  const int bm   = blockIdx.x * BM;
  const int bn   = blockIdx.y * BN;

  const int lrow = tid >> 1;        // staging: row within tile
  const int lcol = (tid & 1) * 16;  // staging: 0 or 16

  const int fm = lane & 15;  // fragment row/col within 16
  const int kh = lane >> 4;  // K-half selector

  const __bf16* gA = A + (size_t)(bm + lrow) * K + lcol;
  const __bf16* gB = B + (size_t)(bn + lrow) * K + lcol;
  __bf16* lA0 = &sA[0][lrow * LDT + lcol];
  __bf16* lB0 = &sB[0][lrow * LDT + lcol];
  __bf16* lA1 = &sA[1][lrow * LDT + lcol];
  __bf16* lB1 = &sB[1][lrow * LDT + lcol];

#if ASYNC_LDS
#define STAGE16(l, g) async_cp16((l), (g))
#else
#define STAGE16(l, g) (*(v8bf*)(l) = *(const v8bf*)(g))
#endif

#define STAGE_TILE(bufsel, koff)                                    \
  do {                                                              \
    __bf16* la_ = (bufsel) ? lA1 : lA0;                             \
    __bf16* lb_ = (bufsel) ? lB1 : lB0;                             \
    STAGE16(la_, gA + (koff));                                      \
    STAGE16(la_ + 8, gA + (koff) + 8);                              \
    STAGE16(lb_, gB + (koff));                                      \
    STAGE16(lb_ + 8, gB + (koff) + 8);                              \
  } while (0)

  v8f acc[4][2];
  const v8f vzero = {0.f, 0.f, 0.f, 0.f, 0.f, 0.f, 0.f, 0.f};
#pragma unroll
  for (int mi = 0; mi < 4; ++mi)
#pragma unroll
    for (int ni = 0; ni < 2; ++ni) acc[mi][ni] = vzero;

  // prologue: stage tile 0 into buffer 0
  STAGE_TILE(0, 0);

  const int nk = K / BK;
  for (int it = 0; it < nk; ++it) {
    const int buf = it & 1;
    WAIT_ASYNC();        // my async writes for this buffer are done
    __syncthreads();     // everyone's writes visible; prev reads of other buf done

    // overlap: stage next K-tile into the other buffer while we compute
    if (it + 1 < nk) STAGE_TILE(buf ^ 1, (it + 1) * BK);

    const __bf16* cA = &sA[buf][0];
    const __bf16* cB = &sB[buf][0];

    // A fragment (16x32 bf16): lane fm holds K = kh*8+0..7 and K = 16+kh*8+0..7
    v16bf af[4];
#pragma unroll
    for (int mi = 0; mi < 4; ++mi) {
      const __bf16* p = &cA[(wm * 64 + mi * 16 + fm) * LDT + kh * 8];
      v8bf lo = *(const v8bf*)p;
      v8bf hi = *(const v8bf*)(p + 16);
      af[mi] = __builtin_shufflevector(lo, hi, 0, 1, 2, 3, 4, 5, 6, 7,
                                       8, 9, 10, 11, 12, 13, 14, 15);
    }
    // B fragment (32x16 bf16): lane fm (=N col) holds K = kh*8+0..7 and K = 16+kh*8+0..7
    v16bf bfrag[2];
#pragma unroll
    for (int ni = 0; ni < 2; ++ni) {
      const __bf16* p = &cB[(wn * 32 + ni * 16 + fm) * LDT + kh * 8];
      v8bf lo = *(const v8bf*)p;
      v8bf hi = *(const v8bf*)(p + 16);
      bfrag[ni] = __builtin_shufflevector(lo, hi, 0, 1, 2, 3, 4, 5, 6, 7,
                                          8, 9, 10, 11, 12, 13, 14, 15);
    }

#pragma unroll
    for (int mi = 0; mi < 4; ++mi)
#pragma unroll
      for (int ni = 0; ni < 2; ++ni)
        acc[mi][ni] = __builtin_amdgcn_wmma_f32_16x16x32_bf16(
            false, af[mi], false, bfrag[ni], (short)0, acc[mi][ni], false, false);

    __syncthreads();  // done reading this buffer before it is re-staged
  }
#undef STAGE16
#undef STAGE_TILE

  // epilogue: C layout is VGPR v -> M = v + 8*(lane>=16), lane&15 -> N
#pragma unroll
  for (int mi = 0; mi < 4; ++mi) {
#pragma unroll
    for (int ni = 0; ni < 2; ++ni) {
#pragma unroll
      for (int v = 0; v < 8; ++v) {
        int row = bm + wm * 64 + mi * 16 + v + (kh << 3);
        int col = bn + wn * 32 + ni * 16 + fm;
        float x = acc[mi][ni][v];
        if (EPI == EPI_BIAS_RELU) x = fmaxf(x + bias[row], 0.f);
        if (EPI == EPI_SIM) x = (x > SIM_THRESH && row != col) ? x : 0.f;
        C[(size_t)row * N + col] = (__bf16)x;
      }
    }
  }
}

// ---------------------------------------------------------------------------
// mean over the contiguous N axis of hT[D, N]: out[j] = mean_i hT[j, i]
// ---------------------------------------------------------------------------
__global__ __launch_bounds__(256) void mean_kernel(const __bf16* __restrict__ hT,
                                                   float* __restrict__ out) {
  __shared__ float red[256];
  const int j = blockIdx.x;
  const int tid = threadIdx.x;
  const __bf16* p = hT + (size_t)j * 4096;
  float s = 0.f;
  for (int i = tid; i < 4096; i += 256) s += (float)p[i];
  red[tid] = s;
  __syncthreads();
  for (int st = 128; st > 0; st >>= 1) {
    if (tid < st) red[tid] += red[tid + st];
    __syncthreads();
  }
  if (tid == 0) out[j] = red[0] * (1.f / 4096.f);
}

// ---------------------------------------------------------------------------
extern "C" void kernel_launch(void* const* d_in, const int* in_sizes, int n_in,
                              void* d_out, int out_size, void* d_ws, size_t ws_size,
                              hipStream_t stream) {
  const float* emb = (const float*)d_in[0];
  const float* W1  = (const float*)d_in[1];
  const float* b1  = (const float*)d_in[2];
  const float* W2  = (const float*)d_in[3];
  const float* b2  = (const float*)d_in[4];
  const float* W3  = (const float*)d_in[5];
  const float* b3  = (const float*)d_in[6];
  float* out = (float*)d_out;

  const int Bb = 8, Nn = 4096, D = 768;

  char* ws = (char*)d_ws;
  size_t off = 0;
  auto alloc = [&](size_t bytes) {
    char* p = ws + off;
    off = (off + bytes + 255) & ~(size_t)255;
    return p;
  };
  __bf16* adj = (__bf16*)alloc((size_t)Nn * Nn * 2);  // 32 MB: L2-resident
  __bf16* nrm = (__bf16*)alloc((size_t)Nn * D * 2);
  __bf16* hTa = (__bf16*)alloc((size_t)D * Nn * 2);
  __bf16* hTb = (__bf16*)alloc((size_t)D * Nn * 2);
  __bf16* t   = (__bf16*)alloc((size_t)Nn * D * 2);
  __bf16* Wb  = (__bf16*)alloc((size_t)3 * D * D * 2);

  convw_kernel<<<(D * D + 255) / 256, 256, 0, stream>>>(W1, W2, W3, Wb);

  const float* biases[3] = {b1, b2, b3};

  for (int b = 0; b < Bb; ++b) {
    const float* e = emb + (size_t)b * Nn * D;
    prep_kernel<<<Nn, 256, 0, stream>>>(e, nrm, hTa);

    // adj = threshold(nrm @ nrm^T), zero diagonal   [4096 x 4096], K = 768
    gemm_nt_bf16<EPI_SIM><<<dim3(Nn / 128, Nn / 128), 256, 0, stream>>>(
        nrm, nrm, adj, nullptr, Nn, Nn, D);

    __bf16* cur = hTa;
    __bf16* nxt = hTb;
    for (int l = 0; l < 3; ++l) {
      // t[N,D] = adj @ h  == NT(adj[N,N], hT[D,N])     K = 4096
      gemm_nt_bf16<EPI_PLAIN><<<dim3(Nn / 128, D / 128), 256, 0, stream>>>(
          adj, cur, t, nullptr, Nn, D, Nn);
      // hT_next[D,N] = relu(W @ t^T + b) == NT(W[D,D], t[N,D]) + bias, relu
      gemm_nt_bf16<EPI_BIAS_RELU><<<dim3(D / 128, Nn / 128), 256, 0, stream>>>(
          Wb + (size_t)l * D * D, t, nxt, biases[l], D, Nn, D);
      __bf16* tmp = cur; cur = nxt; nxt = tmp;
    }
    mean_kernel<<<D, 256, 0, stream>>>(cur, out + (size_t)b * D);
  }
}
